// FastSpeech2_68779606278253
// MI455X (gfx1250) — matrix operand order repaired and checked by reading, not compiled, not run on hardware
//
#include <hip/hip_runtime.h>
#include <hip/hip_bf16.h>
#include <cstdint>
#include <cstddef>

typedef __attribute__((ext_vector_type(16))) _Float16 v16h;
typedef __attribute__((ext_vector_type(8)))  _Float16 v8h;
typedef __attribute__((ext_vector_type(8)))  float    v8f;

// ---------------- model dims ----------------
constexpr int kB = 16, kS = 512, kT = 2048;
constexpr int kD = 768, kH = 4, kDH = 192, kDFF = 1024, kLayers = 4;
constexpr int kDBERT = 448;
constexpr int kMEL = 80, kDECH = 256;
constexpr int kTok    = kB * kS;    // 8192 tokens
constexpr int kFrames = kB * kT;    // 32768 mel frames

// ---------------- workspace layout (bytes; all chunks 256B-aligned) ----------------
constexpr size_t OFF_XF32  = 0;
constexpr size_t OFF_XF16  = OFF_XF32  + (size_t)kTok * kD * 4;
constexpr size_t OFF_WQKV  = OFF_XF16  + (size_t)kTok * kD * 2;
constexpr size_t OFF_WO    = OFF_WQKV  + (size_t)kLayers * 3 * kD * kD * 2;
constexpr size_t OFF_FF1   = OFF_WO    + (size_t)kLayers * kD * kD * 2;
constexpr size_t OFF_FF2   = OFF_FF1   + (size_t)kLayers * kDFF * kD * 2;
constexpr size_t OFF_DPW   = OFF_FF2   + (size_t)kLayers * kD * kDFF * 2;
constexpr size_t OFF_DEC1  = OFF_DPW   + (size_t)2 * 3 * kD * kD * 2;
constexpr size_t OFF_DEC2  = OFF_DEC1  + (size_t)kDECH * kD * 2;
constexpr size_t OFF_BERTP = OFF_DEC2  + (size_t)kMEL * kDECH * 2;
constexpr size_t OFF_QKV16 = OFF_BERTP + (size_t)kB * kDBERT * 4;
constexpr size_t OFF_SCORE = OFF_QKV16 + (size_t)kTok * 3 * kD * 2;
constexpr size_t OFF_P16   = OFF_SCORE + (size_t)kB * kH * kS * kS * 4;
constexpr size_t OFF_O16   = OFF_P16   + (size_t)kB * kH * kS * kS * 2;
constexpr size_t OFF_TMP32 = OFF_O16   + (size_t)kTok * kD * 2;
constexpr size_t OFF_FFH16 = OFF_TMP32 + (size_t)kTok * kD * 4;
// reuse inside / after encoder:
constexpr size_t OFF_VT16  = OFF_TMP32;                          // f16 [B,H,DH,S] (dead region during attention)
constexpr size_t OFF_XPAD0 = OFF_QKV16;                          // f16 [B,S+2,D]
constexpr size_t OFF_XPAD1 = OFF_XPAD0 + (size_t)kB*(kS+2)*kD*2; // f16 [B,S+2,D]
constexpr size_t OFF_CONV  = OFF_TMP32;                          // f32 conv accum [B,S,D]
constexpr size_t OFF_H2    = OFF_O16;                            // f16 dp hidden2 [B,S,D]
constexpr size_t OFF_CUM   = OFF_FFH16;                          // int cumsum [B,S]
constexpr size_t OFF_XE16  = OFF_SCORE;                          // f16 expanded [B,T,D]
constexpr size_t OFF_DECH  = OFF_P16;                            // f16 decoder hidden

// d_out float offsets
constexpr size_t OUT_MEL = 0;
constexpr size_t OUT_DUR = (size_t)kFrames * kMEL;
constexpr size_t OUT_PDL = OUT_DUR + (size_t)kTok;

// ===================================================================
// WMMA GEMM:  C[M,N] = act( A[M,K] * Wt[N,K]^T + bias (+ C) )
// One wave computes a (16*MT) x (16*NT) tile. N % (16*NT) == 0 and
// M % (16*MT) == 0 are guaranteed by the dispatcher -> no divergent
// guards anywhere near v_wmma (EXEC stays all-ones).
// ===================================================================
template <int MT, int NT>
__global__ __launch_bounds__(256) void gemm_wmma_t(
    const _Float16* __restrict__ A, int lda,
    const _Float16* __restrict__ W, int ldw,
    const float* __restrict__ bias,
    float* __restrict__ Cf, _Float16* __restrict__ Ch, int ldc,
    int M, int N, int K, int accum, int relu)
{
  const int wave = threadIdx.x >> 5;
  const int lane = threadIdx.x & 31;
  const int ntiles = N / (16 * NT);
  const int mtiles = M / (16 * MT);
  const int tid = blockIdx.x * 8 + wave;
  if (tid >= mtiles * ntiles) return;
  const int mt = tid / ntiles, nt = tid - mt * ntiles;
  const int row0 = mt * 16 * MT;
  const int col0 = nt * 16 * NT;
  const int l16 = lane & 15;      // row (A) or col (B/C) within 16-tile
  const int kh  = lane >> 4;      // lane-half selects K slice

  v8f acc[MT][NT] = {};
  const _Float16* arow[MT];
#pragma unroll
  for (int m = 0; m < MT; ++m)
    arow[m] = A + (size_t)(row0 + m * 16 + l16) * lda;
  const _Float16* wrow[NT];
#pragma unroll
  for (int t = 0; t < NT; ++t)
    wrow[t] = W + (size_t)(col0 + t * 16 + l16) * ldw + kh * 16;

#pragma unroll 2
  for (int k0 = 0; k0 < K; k0 += 32) {
    // A fragment (16x32 f16): half 0 holds K {0..15}, half 1 {16..31}
    v16h a[MT];
#pragma unroll
    for (int m = 0; m < MT; ++m) {
      v8h lo = *(const v8h*)(arow[m] + k0 + kh * 16);
      v8h hi = *(const v8h*)(arow[m] + k0 + kh * 16 + 8);
#pragma unroll
      for (int j = 0; j < 8; ++j) { a[m][j] = lo[j]; a[m][8 + j] = hi[j]; }
      __builtin_prefetch(arow[m] + k0 + 128, 0, 0);
    }
    // B fragment: lane holds column n=l16, 16 contiguous K values (32B load)
    v16h b[NT];
#pragma unroll
    for (int t = 0; t < NT; ++t) b[t] = *(const v16h*)(wrow[t] + k0);
#pragma unroll
    for (int m = 0; m < MT; ++m)
#pragma unroll
      for (int t = 0; t < NT; ++t)
        acc[m][t] = __builtin_amdgcn_wmma_f32_16x16x32_f16(
            false, a[m], false, b[t], (short)0, acc[m][t], false, false);
  }

  // epilogue: C layout — vgpr r, lane-half kh -> row r + 8*kh ; col = lane&15
#pragma unroll
  for (int m = 0; m < MT; ++m)
#pragma unroll
    for (int t = 0; t < NT; ++t) {
      const int col = col0 + t * 16 + l16;
      const float bv = bias ? bias[col] : 0.f;
#pragma unroll
      for (int r = 0; r < 8; ++r) {
        const int row = row0 + m * 16 + r + 8 * kh;
        const size_t off = (size_t)row * ldc + col;
        float v = acc[m][t][r] + bv;
        if (accum && Cf) v += Cf[off];
        if (relu) v = v > 0.f ? v : 0.f;
        if (Cf) Cf[off] = v;
        if (Ch) Ch[off] = (_Float16)v;
      }
    }
}

// ---------------- block reductions ----------------
__device__ __forceinline__ float block_sum(float v, float* buf) {
  const int tid = threadIdx.x;
  buf[tid] = v; __syncthreads();
  for (int s = 128; s > 0; s >>= 1) { if (tid < s) buf[tid] += buf[tid + s]; __syncthreads(); }
  float r = buf[0]; __syncthreads(); return r;
}
__device__ __forceinline__ float block_max(float v, float* buf) {
  const int tid = threadIdx.x;
  buf[tid] = v; __syncthreads();
  for (int s = 128; s > 0; s >>= 1) { if (tid < s) buf[tid] = fmaxf(buf[tid], buf[tid + s]); __syncthreads(); }
  float r = buf[0]; __syncthreads(); return r;
}

// ---------------- small kernels ----------------
__global__ void cvt_f16_kernel(const float* __restrict__ s, _Float16* __restrict__ d, int n) {
  int i = blockIdx.x * 256 + threadIdx.x;
  if (i < n) d[i] = (_Float16)s[i];
}

// dp conv weights [O,I,3] -> 3 contiguous f16 tap matrices [O,I]
__global__ void extract_taps_kernel(const float* __restrict__ s, _Float16* __restrict__ d, int OI) {
  int i = blockIdx.x * 256 + threadIdx.x;
  if (i < OI) {
    d[i]          = (_Float16)s[(size_t)i * 3 + 0];
    d[OI + i]     = (_Float16)s[(size_t)i * 3 + 1];
    d[2 * OI + i] = (_Float16)s[(size_t)i * 3 + 2];
  }
}

// vt[((b*H+h)*DH + d), s] = V[b,s,h,d]  (makes P@V a contiguous Wt-GEMM)
__global__ void transpose_v_kernel(const _Float16* __restrict__ qkv, _Float16* __restrict__ vt) {
  const int row = blockIdx.x;               // (b*H+h)*DH + d
  const int d = row % kDH; const int bh = row / kDH;
  const int h = bh % kH,   b  = bh / kH;
  const _Float16* src = qkv + (size_t)b * kS * 3 * kD + 2 * kD + (size_t)h * kDH + d;
  _Float16* dst = vt + (size_t)row * kS;
  for (int s = threadIdx.x; s < kS; s += 256) dst[s] = src[(size_t)s * 3 * kD];
}

__global__ void bert_proj_kernel(const float* __restrict__ bert, const float* __restrict__ W,
                                 const float* __restrict__ bb, float* __restrict__ out) {
  int o = blockIdx.x * 256 + threadIdx.x;
  if (o >= kB * kDBERT) return;
  int b = o / kDBERT, c = o - b * kDBERT;
  const float* x = bert + (size_t)b * kDBERT;
  const float* w = W + (size_t)c * kDBERT;
  float s = bb[c];
  for (int i = 0; i < kDBERT; ++i) s += x[i] * w[i];
  out[o] = s;
}

// x = concat(ph, sp, bert_proj) + positional encoding ; writes f32 + f16
__global__ __launch_bounds__(256) void embed_kernel(
    const int* __restrict__ ph, const int* __restrict__ spk, const float* __restrict__ bertp,
    const float* __restrict__ eph, const float* __restrict__ esp,
    float* __restrict__ xf, _Float16* __restrict__ xh)
{
  const int row = blockIdx.x;            // b*S + s
  const int b = row / kS, s = row - b * kS;
  const int pid = ph[row], sid = spk[b];
  for (int d = threadIdx.x; d < kD; d += 256) {
    float v;
    if (d < 256)       v = eph[(size_t)pid * 256 + d];
    else if (d < 320)  v = esp[(size_t)sid * 64 + (d - 256)];
    else               v = bertp[(size_t)b * kDBERT + (d - 320)];
    const int j = d >> 1;
    const float ang = (float)s * __expf(-(float)(2 * j) * (9.210340372f / (float)kD));
    v += (d & 1) ? __cosf(ang) : __sinf(ang);
    const size_t off = (size_t)row * kD + d;
    xf[off] = v; xh[off] = (_Float16)v;
  }
}

// softmax over rows of length 512 (scaled), f32 in -> f16 out
__global__ __launch_bounds__(256) void softmax_kernel(const float* __restrict__ S_,
                                                      _Float16* __restrict__ P, float scale) {
  __shared__ float buf[256];
  const size_t row = blockIdx.x;
  const float* src = S_ + row * kS;
  float a0 = src[threadIdx.x] * scale;
  float a1 = src[threadIdx.x + 256] * scale;
  float m = block_max(fmaxf(a0, a1), buf);
  float e0 = __expf(a0 - m), e1 = __expf(a1 - m);
  float inv = 1.f / block_sum(e0 + e1, buf);
  P[row * kS + threadIdx.x]       = (_Float16)(e0 * inv);
  P[row * kS + threadIdx.x + 256] = (_Float16)(e1 * inv);
}

// x = LayerNorm(x + t) ; updates f32 and f16 copies (D=768, 3 elems/thread)
__global__ __launch_bounds__(256) void add_ln_kernel(
    float* __restrict__ x, _Float16* __restrict__ xh, const float* __restrict__ t,
    const float* __restrict__ g, const float* __restrict__ bta)
{
  __shared__ float buf[256];
  const size_t row = blockIdx.x;
  float v[3]; float s = 0.f;
#pragma unroll
  for (int i = 0; i < 3; ++i) {
    int idx = threadIdx.x + i * 256;
    v[i] = x[row * kD + idx] + t[row * kD + idx];
    s += v[i];
  }
  const float mean = block_sum(s, buf) * (1.f / kD);
  float s2 = 0.f;
#pragma unroll
  for (int i = 0; i < 3; ++i) { float d = v[i] - mean; s2 += d * d; }
  const float rstd = rsqrtf(block_sum(s2, buf) * (1.f / kD) + 1e-5f);
#pragma unroll
  for (int i = 0; i < 3; ++i) {
    int idx = threadIdx.x + i * 256;
    float o = (v[i] - mean) * rstd * g[idx] + bta[idx];
    x[row * kD + idx] = o; xh[row * kD + idx] = (_Float16)o;
  }
}

// dst = LayerNorm(relu(src)) -> f16; pad==1 maps row (b,s) into padded [B,S+2,D] interior
__global__ __launch_bounds__(256) void relu_ln_kernel(
    const float* __restrict__ src, const float* __restrict__ g, const float* __restrict__ bta,
    _Float16* __restrict__ dst, int pad)
{
  __shared__ float buf[256];
  const int row = blockIdx.x;
  const int b = row / kS, s = row - b * kS;
  const size_t drow = pad ? ((size_t)b * (kS + 2) + s + 1) : (size_t)row;
  float v[3]; float sum = 0.f;
#pragma unroll
  for (int i = 0; i < 3; ++i) {
    int idx = threadIdx.x + i * 256;
    float a = src[(size_t)row * kD + idx];
    v[i] = a > 0.f ? a : 0.f;
    sum += v[i];
  }
  const float mean = block_sum(sum, buf) * (1.f / kD);
  float s2 = 0.f;
#pragma unroll
  for (int i = 0; i < 3; ++i) { float d = v[i] - mean; s2 += d * d; }
  const float rstd = rsqrtf(block_sum(s2, buf) * (1.f / kD) + 1e-5f);
#pragma unroll
  for (int i = 0; i < 3; ++i) {
    int idx = threadIdx.x + i * 256;
    dst[drow * kD + idx] = (_Float16)((v[i] - mean) * rstd * g[idx] + bta[idx]);
  }
}

// build padded conv input from encoder f16 activations
__global__ void build_xpad_kernel(const _Float16* __restrict__ xh, _Float16* __restrict__ xpad) {
  const int row = blockIdx.x;                 // b*(S+2)+r
  const int b = row / (kS + 2), r = row - b * (kS + 2);
  const bool edge = (r == 0) || (r == kS + 1);
  for (int d = threadIdx.x; d < kD; d += 256)
    xpad[(size_t)row * kD + d] = edge ? (_Float16)0.f : xh[((size_t)b * kS + r - 1) * kD + d];
}

__global__ void zero_pad_edges_kernel(_Float16* __restrict__ xpad) {
  const int i = blockIdx.x;                   // 2*B blocks
  const int b = i >> 1;
  const size_t row = (size_t)b * (kS + 2) + ((i & 1) ? (kS + 1) : 0);
  for (int d = threadIdx.x; d < kD; d += 256) xpad[row * kD + d] = (_Float16)0.f;
}

// pred_log_dur[b,s] = dot(h2[b,s,:], wo) + bo
__global__ __launch_bounds__(256) void dp_out_kernel(const _Float16* __restrict__ h,
                                                     const float* __restrict__ wo,
                                                     const float* __restrict__ bo,
                                                     float* __restrict__ out) {
  __shared__ float buf[256];
  const size_t row = blockIdx.x;
  float s = 0.f;
  for (int d = threadIdx.x; d < kD; d += 256) s += (float)h[row * kD + d] * wo[d];
  float tot = block_sum(s, buf);
  if (threadIdx.x == 0) out[row] = tot + bo[0];
}

__global__ void cumsum_kernel(const int* __restrict__ dur, int* __restrict__ cum) {
  if (threadIdx.x == 0) {
    const int b = blockIdx.x;
    int a = 0;
    for (int s = 0; s < kS; ++s) { a += dur[b * kS + s]; cum[b * kS + s] = a; }
  }
}

// length regulator: searchsorted(cum, t, right) gather, zero invalid frames
__global__ __launch_bounds__(256) void gather_kernel(const _Float16* __restrict__ xh,
                                                     const int* __restrict__ cum,
                                                     const int* __restrict__ mellen,
                                                     _Float16* __restrict__ xe) {
  const int row = blockIdx.x;                 // b*T + t
  const int b = row / kT, t = row - b * kT;
  const int* c = cum + b * kS;
  const bool valid = (t < c[kS - 1]) && (t < mellen[b]);
  int lo = 0, hi = kS;
  while (lo < hi) { int mid = (lo + hi) >> 1; if (c[mid] <= t) lo = mid + 1; else hi = mid; }
  const int idx = lo < kS - 1 ? lo : kS - 1;
  const _Float16* srcrow = xh + ((size_t)b * kS + idx) * kD;
  for (int d = threadIdx.x; d < kD; d += 256)
    xe[(size_t)row * kD + d] = valid ? srcrow[d] : (_Float16)0.f;
}

__global__ void dur_copy_kernel(const int* __restrict__ dur, float* __restrict__ out) {
  int i = blockIdx.x * 256 + threadIdx.x;
  if (i < kTok) out[i] = (float)dur[i];
}

// ---------------- host helpers ----------------
static inline void launch_gemm(hipStream_t st, const _Float16* A, int lda,
                               const _Float16* W, int ldw, const float* bias,
                               float* Cf, _Float16* Ch, int ldc,
                               int M, int N, int K, int accum, int relu) {
  // M % 32 == 0 for every call site; N % 64 == 0 except N == 80 (% 80 == 0).
  if (N % 64 == 0) {
    const int tiles = (M / 32) * (N / 64);
    gemm_wmma_t<2, 4><<<(tiles + 7) / 8, 256, 0, st>>>(A, lda, W, ldw, bias, Cf, Ch, ldc, M, N, K, accum, relu);
  } else if (N % 80 == 0) {
    const int tiles = (M / 32) * (N / 80);
    gemm_wmma_t<2, 5><<<(tiles + 7) / 8, 256, 0, st>>>(A, lda, W, ldw, bias, Cf, Ch, ldc, M, N, K, accum, relu);
  } else {
    const int tiles = (M / 32) * (N / 16);
    gemm_wmma_t<2, 1><<<(tiles + 7) / 8, 256, 0, st>>>(A, lda, W, ldw, bias, Cf, Ch, ldc, M, N, K, accum, relu);
  }
}
static inline void launch_cvt(hipStream_t st, const float* s, _Float16* d, int n) {
  cvt_f16_kernel<<<(n + 255) / 256, 256, 0, st>>>(s, d, n);
}

extern "C" void kernel_launch(void* const* d_in, const int* in_sizes, int n_in,
                              void* d_out, int out_size, void* d_ws, size_t ws_size,
                              hipStream_t stream) {
  (void)in_sizes; (void)n_in; (void)out_size; (void)ws_size;
  const int*   ph     = (const int*)d_in[0];
  const int*   spk    = (const int*)d_in[1];
  const float* bert   = (const float*)d_in[2];
  const int*   dur    = (const int*)d_in[3];
  const int*   mellen = (const int*)d_in[4];
  const float* emb_ph = (const float*)d_in[5];
  const float* emb_sp = (const float*)d_in[6];
  const float* W_bert = (const float*)d_in[7];
  const float* b_bert = (const float*)d_in[8];
  const float* Wqkv   = (const float*)d_in[9];
  const float* bqkv   = (const float*)d_in[10];
  const float* Wo     = (const float*)d_in[11];
  const float* bo     = (const float*)d_in[12];
  const float* ln1_g  = (const float*)d_in[13];
  const float* ln1_b  = (const float*)d_in[14];
  const float* ff_w1  = (const float*)d_in[15];
  const float* ff_b1  = (const float*)d_in[16];
  const float* ff_w2  = (const float*)d_in[17];
  const float* ff_b2  = (const float*)d_in[18];
  const float* ln2_g  = (const float*)d_in[19];
  const float* ln2_b  = (const float*)d_in[20];
  const float* dp_w1  = (const float*)d_in[21];
  const float* dp_b1  = (const float*)d_in[22];
  const float* dp_g1  = (const float*)d_in[23];
  const float* dp_be1 = (const float*)d_in[24];
  const float* dp_w2  = (const float*)d_in[25];
  const float* dp_b2  = (const float*)d_in[26];
  const float* dp_g2  = (const float*)d_in[27];
  const float* dp_be2 = (const float*)d_in[28];
  const float* dp_wo  = (const float*)d_in[29];
  const float* dp_bo  = (const float*)d_in[30];
  const float* dec_w1 = (const float*)d_in[31];
  const float* dec_b1 = (const float*)d_in[32];
  const float* dec_w2 = (const float*)d_in[33];
  const float* dec_b2 = (const float*)d_in[34];

  char* ws = (char*)d_ws;
  float*    xf     = (float*)(ws + OFF_XF32);
  _Float16* xh     = (_Float16*)(ws + OFF_XF16);
  _Float16* wqkv16 = (_Float16*)(ws + OFF_WQKV);
  _Float16* wo16   = (_Float16*)(ws + OFF_WO);
  _Float16* ff116  = (_Float16*)(ws + OFF_FF1);
  _Float16* ff216  = (_Float16*)(ws + OFF_FF2);
  _Float16* dpw16  = (_Float16*)(ws + OFF_DPW);
  _Float16* dec116 = (_Float16*)(ws + OFF_DEC1);
  _Float16* dec216 = (_Float16*)(ws + OFF_DEC2);
  float*    bertp  = (float*)(ws + OFF_BERTP);
  _Float16* qkv16  = (_Float16*)(ws + OFF_QKV16);
  float*    score  = (float*)(ws + OFF_SCORE);
  _Float16* p16    = (_Float16*)(ws + OFF_P16);
  _Float16* o16    = (_Float16*)(ws + OFF_O16);
  float*    tmp32  = (float*)(ws + OFF_TMP32);
  _Float16* vt16   = (_Float16*)(ws + OFF_VT16);
  _Float16* ffh16  = (_Float16*)(ws + OFF_FFH16);
  _Float16* xpad0  = (_Float16*)(ws + OFF_XPAD0);
  _Float16* xpad1  = (_Float16*)(ws + OFF_XPAD1);
  float*    convf  = (float*)(ws + OFF_CONV);
  _Float16* h2f    = (_Float16*)(ws + OFF_H2);
  int*      cum    = (int*)(ws + OFF_CUM);
  _Float16* xe16   = (_Float16*)(ws + OFF_XE16);
  _Float16* dech   = (_Float16*)(ws + OFF_DECH);
  float*    out    = (float*)d_out;

  // ---- convert weights to f16 (deterministic, every call) ----
  launch_cvt(stream, Wqkv,   wqkv16, kLayers * 3 * kD * kD);
  launch_cvt(stream, Wo,     wo16,   kLayers * kD * kD);
  launch_cvt(stream, ff_w1,  ff116,  kLayers * kDFF * kD);
  launch_cvt(stream, ff_w2,  ff216,  kLayers * kD * kDFF);
  launch_cvt(stream, dec_w1, dec116, kDECH * kD);
  launch_cvt(stream, dec_w2, dec216, kMEL * kDECH);
  extract_taps_kernel<<<(kD * kD + 255) / 256, 256, 0, stream>>>(dp_w1, dpw16, kD * kD);
  extract_taps_kernel<<<(kD * kD + 255) / 256, 256, 0, stream>>>(dp_w2, dpw16 + (size_t)3 * kD * kD, kD * kD);

  // ---- embedding + positional encoding ----
  bert_proj_kernel<<<(kB * kDBERT + 255) / 256, 256, 0, stream>>>(bert, W_bert, b_bert, bertp);
  embed_kernel<<<kTok, 256, 0, stream>>>(ph, spk, bertp, emb_ph, emb_sp, xf, xh);

  // ---- transformer encoder ----
  const float scale = 0.07216878364870323f;   // 1/sqrt(192)
  for (int l = 0; l < kLayers; ++l) {
    // qkv = x @ Wqkv^T + b   -> f16 [Tok, 3D]
    launch_gemm(stream, xh, kD, wqkv16 + (size_t)l * 3 * kD * kD, kD,
                bqkv + (size_t)l * 3 * kD, nullptr, qkv16, 3 * kD, kTok, 3 * kD, kD, 0, 0);
    // scores = Q K^T per (b,h)
    for (int b = 0; b < kB; ++b)
      for (int h = 0; h < kH; ++h) {
        const _Float16* qb = qkv16 + (size_t)b * kS * 3 * kD + (size_t)h * kDH;
        const _Float16* kb = qb + kD;
        float* sc = score + (size_t)(b * kH + h) * kS * kS;
        launch_gemm(stream, qb, 3 * kD, kb, 3 * kD, nullptr,
                    sc, nullptr, kS, kS, kS, kDH, 0, 0);
      }
    softmax_kernel<<<kB * kH * kS, 256, 0, stream>>>(score, p16, scale);
    // V transpose -> [B,H,DH,S], then O = P @ V as a contiguous Wt-GEMM
    transpose_v_kernel<<<kB * kH * kDH, 256, 0, stream>>>(qkv16, vt16);
    for (int b = 0; b < kB; ++b)
      for (int h = 0; h < kH; ++h) {
        const _Float16* pb = p16 + (size_t)(b * kH + h) * kS * kS;
        const _Float16* vtb = vt16 + (size_t)(b * kH + h) * kDH * kS;
        _Float16* ob = o16 + (size_t)b * kS * kD + (size_t)h * kDH;
        launch_gemm(stream, pb, kS, vtb, kS, nullptr,
                    nullptr, ob, kD, kS, kDH, kS, 0, 0);
      }
    // out projection + residual + LN1
    launch_gemm(stream, o16, kD, wo16 + (size_t)l * kD * kD, kD,
                bo + (size_t)l * kD, tmp32, nullptr, kD, kTok, kD, kD, 0, 0);
    add_ln_kernel<<<kTok, 256, 0, stream>>>(xf, xh, tmp32,
                                            ln1_g + (size_t)l * kD, ln1_b + (size_t)l * kD);
    // FFN
    launch_gemm(stream, xh, kD, ff116 + (size_t)l * kDFF * kD, kD,
                ff_b1 + (size_t)l * kDFF, nullptr, ffh16, kDFF, kTok, kDFF, kD, 0, 1);
    launch_gemm(stream, ffh16, kDFF, ff216 + (size_t)l * kD * kDFF, kDFF,
                ff_b2 + (size_t)l * kD, tmp32, nullptr, kD, kTok, kD, kDFF, 0, 0);
    add_ln_kernel<<<kTok, 256, 0, stream>>>(xf, xh, tmp32,
                                            ln2_g + (size_t)l * kD, ln2_b + (size_t)l * kD);
  }

  // ---- duration predictor: conv(k=3) as 3 shifted tap-GEMMs ----
  build_xpad_kernel<<<kB * (kS + 2), 256, 0, stream>>>(xh, xpad0);
  for (int b = 0; b < kB; ++b)
    for (int t = 0; t < 3; ++t)
      launch_gemm(stream, xpad0 + ((size_t)b * (kS + 2) + t) * kD, kD,
                  dpw16 + (size_t)t * kD * kD, kD,
                  t == 0 ? dp_b1 : nullptr,
                  convf + (size_t)b * kS * kD, nullptr, kD, kS, kD, kD, t > 0, 0);
  relu_ln_kernel<<<kTok, 256, 0, stream>>>(convf, dp_g1, dp_be1, xpad1, 1);
  zero_pad_edges_kernel<<<2 * kB, 256, 0, stream>>>(xpad1);
  for (int b = 0; b < kB; ++b)
    for (int t = 0; t < 3; ++t)
      launch_gemm(stream, xpad1 + ((size_t)b * (kS + 2) + t) * kD, kD,
                  dpw16 + (size_t)(3 + t) * kD * kD, kD,
                  t == 0 ? dp_b2 : nullptr,
                  convf + (size_t)b * kS * kD, nullptr, kD, kS, kD, kD, t > 0, 0);
  relu_ln_kernel<<<kTok, 256, 0, stream>>>(convf, dp_g2, dp_be2, h2f, 0);
  dp_out_kernel<<<kTok, 256, 0, stream>>>(h2f, dp_wo, dp_bo, out + OUT_PDL);
  dur_copy_kernel<<<(kTok + 255) / 256, 256, 0, stream>>>(dur, out + OUT_DUR);

  // ---- length regulator ----
  cumsum_kernel<<<kB, 32, 0, stream>>>(dur, cum);
  gather_kernel<<<kFrames, 256, 0, stream>>>(xh, cum, mellen, xe16);

  // ---- decoder MLP -> mel ----
  launch_gemm(stream, xe16, kD, dec116, kD, dec_b1,
              nullptr, dech, kDECH, kFrames, kDECH, kD, 0, 1);
  launch_gemm(stream, dech, kDECH, dec216, kDECH, dec_b2,
              out + OUT_MEL, nullptr, kMEL, kFrames, kMEL, kDECH, 0, 0);
}
